// RNN_ANN_28905129902335
// MI455X (gfx1250) — matrix operand order repaired and checked by reading, not compiled, not run on hardware
//
#include <hip/hip_runtime.h>

// Elman RNN for MI455X (gfx1250, wave32, WMMA f16->f32).
// Strategy: batch-split recurrence (no inter-WG sync), h state in LDS in
// WMMA A-swizzle, half of W_hh cached in LDS, half streamed from L2.
// x_proj is software-pipelined: t+1 in registers, t+2 prefetched to cache.

typedef __attribute__((ext_vector_type(16))) _Float16 v16h;
typedef __attribute__((ext_vector_type(8)))  _Float16 v8h;
typedef __attribute__((ext_vector_type(8)))  float    v8f;

#define T_STEPS 512
#define BATCH   256
#define DIN     256
#define DH      512

// ---- WMMA operand loaders -------------------------------------------------
// A operand (16x32, 16-bit), source row-major [M x K]:
//   lane<16 : K 0..7 (v0-3), 16..23 (v4-7);  lane>=16 : K 8..15, 24..31
__device__ __forceinline__ v16h load_opA_f32(const float* p, int ld, int row0,
                                             int k0, int lane) {
  const float* rp = p + (size_t)(row0 + (lane & 15)) * ld + k0 + ((lane & 16) ? 8 : 0);
  float4 a = *(const float4*)rp;
  float4 b = *(const float4*)(rp + 4);
  float4 c = *(const float4*)(rp + 16);
  float4 d = *(const float4*)(rp + 20);
  v16h o;
  o[0]  = (_Float16)a.x; o[1]  = (_Float16)a.y; o[2]  = (_Float16)a.z; o[3]  = (_Float16)a.w;
  o[4]  = (_Float16)b.x; o[5]  = (_Float16)b.y; o[6]  = (_Float16)b.z; o[7]  = (_Float16)b.w;
  o[8]  = (_Float16)c.x; o[9]  = (_Float16)c.y; o[10] = (_Float16)c.z; o[11] = (_Float16)c.w;
  o[12] = (_Float16)d.x; o[13] = (_Float16)d.y; o[14] = (_Float16)d.z; o[15] = (_Float16)d.w;
  return o;
}

// B operand (32x16, 16-bit), source = W^T row-major [N x K]:
//   lane<16 : K 0..7, 16..23;  lane>=16 : K 8..15, 24..31
__device__ __forceinline__ v16h load_opB_f16(const _Float16* pT, int ld, int n0,
                                             int k0, int lane) {
  const _Float16* rp = pT + (size_t)(n0 + (lane & 15)) * ld + k0 + ((lane & 16) ? 8 : 0);
  v8h lo = *(const v8h*)rp;         // 16B aligned
  v8h hi = *(const v8h*)(rp + 16);
  return __builtin_shufflevector(lo, hi, 0,1,2,3,4,5,6,7,8,9,10,11,12,13,14,15);
}

// ---- Prep: transpose + f32->f16 of W_hh and W_ih --------------------------
__global__ __launch_bounds__(256) void k_prep(const float* __restrict__ W_ih,
                                              const float* __restrict__ W_hh,
                                              _Float16* __restrict__ WihT,
                                              _Float16* __restrict__ WhhT) {
  int idx = blockIdx.x * 256 + threadIdx.x;
  if (idx < DH * DH) {                       // WhhT[n][k] = W_hh[k][n]
    int n = idx / DH, k = idx % DH;
    WhhT[(size_t)n * DH + k] = (_Float16)W_hh[(size_t)k * DH + n];
  } else if (idx < DH * DH + DH * DIN) {     // WihT[n][k] = W_ih[k][n]
    int r = idx - DH * DH;
    int n = r / DIN, k = r % DIN;
    WihT[(size_t)n * DIN + k] = (_Float16)W_ih[(size_t)k * DH + n];
  }
}

// ---- Phase 1: x_proj[t,b,h] = sec @ W_ih + (b_ih + b_hh) ------------------
// M = T*B = 131072, K = 256, N = 512. Grid (512, 8), 256 threads (8 waves).
// Wave: 2 m-tiles x 4 n-tiles.
__global__ __launch_bounds__(256) void k_xproj(const float* __restrict__ sec,
                                               const _Float16* __restrict__ WihT,
                                               const float* __restrict__ b_ih,
                                               const float* __restrict__ b_hh,
                                               float* __restrict__ xp) {
  int lane = threadIdx.x & 31, wid = threadIdx.x >> 5;
  int mt0 = (blockIdx.x * 16 + wid * 2) * 16;   // first row of wave's m-tiles
  int nt0 = blockIdx.y * 4;                     // first n-tile index
  v8f acc[2][4] = {};
  for (int kt = 0; kt < DIN / 32; ++kt) {
    int k0 = kt * 32;
    v16h a0 = load_opA_f32(sec, DIN, mt0,      k0, lane);
    v16h a1 = load_opA_f32(sec, DIN, mt0 + 16, k0, lane);
    for (int j = 0; j < 4; ++j) {
      v16h b = load_opB_f16(WihT, DIN, (nt0 + j) * 16, k0, lane);
      acc[0][j] = __builtin_amdgcn_wmma_f32_16x16x32_f16(false, a0, false, b,
                                                         (short)0, acc[0][j], false, false);
      acc[1][j] = __builtin_amdgcn_wmma_f32_16x16x32_f16(false, a1, false, b,
                                                         (short)0, acc[1][j], false, false);
    }
  }
  int hi8 = (lane & 16) ? 8 : 0;
  for (int j = 0; j < 4; ++j) {
    int col = (nt0 + j) * 16 + (lane & 15);
    float bias = b_ih[col] + b_hh[col];        // fold both biases into xp
    for (int i = 0; i < 2; ++i) {
      int rbase = mt0 + i * 16 + hi8;
      for (int v = 0; v < 8; ++v)
        xp[(size_t)(rbase + v) * DH + col] = acc[i][j][v] + bias;
    }
  }
}

// ---- Phase 2: 512 recurrent steps, batch-split over 16 WGs ----------------
// WG g owns batch rows [16g, 16g+16). h state lives in LDS in A-swizzle,
// double buffered. W_hh^T k-tiles 0..7 cached in LDS, 8..15 streamed (L2).
// x_proj pipelined: registers hold t+1's tile, t+2's tile is prefetched.
__global__ __launch_bounds__(256) void k_rnn(const float* __restrict__ xp,
                                             const _Float16* __restrict__ WhhT,
                                             float* __restrict__ hfin) {
  __shared__ __align__(32) _Float16 sW[8][32][32][16];  // 256 KB: [kt][nt][lane][slot]
  __shared__ __align__(32) _Float16 sH[2][16][32][16];  //  32 KB: [buf][kt][lane][slot]
  int lane = threadIdx.x & 31, wid = threadIdx.x >> 5;
  int wg = blockIdx.x;                                  // 0..15
  int hi8 = (lane & 16) ? 8 : 0;

  // h_0 = 0
  for (int i = threadIdx.x; i < 2 * 16 * 32 * 16; i += 256)
    ((_Float16*)sH)[i] = (_Float16)0.0f;
  // pack B-operand tiles for k-tiles 0..7 into LDS (wave w -> k-tile w)
  for (int nt = 0; nt < 32; ++nt) {
    v16h b = load_opB_f16(WhhT, DH, nt * 16, wid * 32, lane);
    *(v16h*)&sW[wid][nt][lane][0] = b;
  }
  __syncthreads();

  // x_proj tile base for this WG at step t: xp + (t*BATCH + wg*16)*DH
  const size_t xp_step = (size_t)BATCH * DH;            // floats per step
  const float* xp_wg = xp + (size_t)wg * 16 * DH;

  // preload x_proj[t=0] into registers
  v8f xnext[4];
  for (int j = 0; j < 4; ++j) {
    int col = (wid * 4 + j) * 16 + (lane & 15);
    for (int v = 0; v < 8; ++v)
      xnext[j][v] = xp_wg[(size_t)(hi8 + v) * DH + col];
  }
  // prefetch x_proj[t=1] tile (32 KB: one 128B line per thread)
  __builtin_prefetch(xp_wg + xp_step + (size_t)threadIdx.x * 32, 0, 1);

  for (int t = 0; t < T_STEPS; ++t) {
    int cur = t & 1, nxt = cur ^ 1;
    // consume pipelined x_proj[t]
    v8f acc[4];
    for (int j = 0; j < 4; ++j) acc[j] = xnext[j];
    // issue loads for x_proj[t+1]; they complete during this step's GEMM
    if (t + 1 < T_STEPS) {
      const float* xpt = xp_wg + (size_t)(t + 1) * xp_step;
      for (int j = 0; j < 4; ++j) {
        int col = (wid * 4 + j) * 16 + (lane & 15);
        for (int v = 0; v < 8; ++v)
          xnext[j][v] = xpt[(size_t)(hi8 + v) * DH + col];
      }
    }
    // prefetch x_proj[t+2] into cache hierarchy
    if (t + 2 < T_STEPS)
      __builtin_prefetch(xp_wg + (size_t)(t + 2) * xp_step +
                         (size_t)threadIdx.x * 32, 0, 1);

    // h(16x512) @ W_hh(512x512): wave handles n-tiles 4w..4w+3
    for (int kt = 0; kt < 16; ++kt) {
      v16h a = *(const v16h*)&sH[cur][kt][lane][0];
      for (int j = 0; j < 4; ++j) {
        int nt = wid * 4 + j;
        v16h b = (kt < 8) ? *(const v16h*)&sW[kt][nt][lane][0]
                          : load_opB_f16(WhhT, DH, nt * 16, kt * 32, lane);
        acc[j] = __builtin_amdgcn_wmma_f32_16x16x32_f16(false, a, false, b,
                                                        (short)0, acc[j], false, false);
      }
    }
    // tanh + scatter into next step's A-swizzle in LDS
    for (int j = 0; j < 4; ++j) {
      int n = (wid * 4 + j) * 16 + (lane & 15);
      int s = n >> 5, kl = n & 31;
      int slot = (kl & 7) | ((kl & 16) ? 8 : 0);
      int lp = (kl & 8) ? 16 : 0;
      for (int v = 0; v < 8; ++v) {
        int m = hi8 + v;
        float hv = tanhf(acc[j][v]);
        sH[nxt][s][m + lp][slot] = (_Float16)hv;
        if (t == T_STEPS - 1)
          hfin[(size_t)(wg * 16 + m) * DH + n] = hv;
      }
    }
    __syncthreads();
  }
}

// ---- Phase 3: out[b] = tanh(h_final @ W_out + b_out) ----------------------
__global__ __launch_bounds__(256) void k_out(const float* __restrict__ hfin,
                                             const float* __restrict__ Wout,
                                             const float* __restrict__ bout,
                                             float* __restrict__ out) {
  int b = threadIdx.x;
  float acc = 0.0f;
  for (int k = 0; k < DH; ++k)
    acc = fmaf(hfin[(size_t)b * DH + k], Wout[k], acc);
  out[b] = tanhf(acc + bout[0]);
}

extern "C" void kernel_launch(void* const* d_in, const int* in_sizes, int n_in,
                              void* d_out, int out_size, void* d_ws, size_t ws_size,
                              hipStream_t stream) {
  (void)in_sizes; (void)n_in; (void)out_size; (void)ws_size;
  const float* sec   = (const float*)d_in[0];
  const float* W_ih  = (const float*)d_in[1];
  const float* b_ih  = (const float*)d_in[2];
  const float* W_hh  = (const float*)d_in[3];
  const float* b_hh  = (const float*)d_in[4];
  const float* W_out = (const float*)d_in[5];
  const float* b_out = (const float*)d_in[6];

  char* ws = (char*)d_ws;
  const size_t XP_BYTES = (size_t)T_STEPS * BATCH * DH * sizeof(float); // 256 MB
  float*    xp   = (float*)ws;
  _Float16* WhhT = (_Float16*)(ws + XP_BYTES);
  _Float16* WihT = (_Float16*)(ws + XP_BYTES + (size_t)DH * DH * 2);
  float*    hfin = (float*)(ws + XP_BYTES + (size_t)DH * DH * 2 + (size_t)DH * DIN * 2);

  k_prep <<<1536, 256, 0, stream>>>(W_ih, W_hh, WihT, WhhT);
  k_xproj<<<dim3(512, 8), 256, 0, stream>>>(sec, WihT, b_ih, b_hh, xp);
  k_rnn  <<<16, 256, 0, stream>>>(xp, WhhT, hfin);
  k_out  <<<1, 256, 0, stream>>>(hfin, W_out, b_out, (float*)d_out);
}
